// ShiftedWindowAttention_54116587929846
// MI455X (gfx1250) — matrix operand order repaired and checked by reading, not compiled, not run on hardware
//
#include <hip/hip_runtime.h>
#include <hip/hip_bf16.h>
#include <cstdint>

typedef _Float16 f16;
typedef _Float16 v16h __attribute__((ext_vector_type(16)));
typedef _Float16 v8h  __attribute__((ext_vector_type(8)));
typedef float    v8f  __attribute__((ext_vector_type(8)));
typedef unsigned int v4u __attribute__((ext_vector_type(4)));
typedef int      v8i  __attribute__((ext_vector_type(8)));
typedef int      v4i  __attribute__((ext_vector_type(4)));

static __device__ __forceinline__ v8f wmma_f16(v16h a, v16h b, v8f c) {
  // codegen-verified: v_wmma_f32_16x16x32_f16
  return __builtin_amdgcn_wmma_f32_16x16x32_f16(false, a, false, b, (short)0, c, false, false);
}

// A-fragment (16x32 f16), contiguous K order: lane half h holds K chunks [16h..16h+7] and [16h+8..16h+15]
static __device__ __forceinline__ v16h afrag(const f16* lo, const f16* hi) {
  v8h l = *(const v8h*)lo;
  v8h h = *(const v8h*)hi;
  v16h a;
#pragma unroll
  for (int i = 0; i < 8; ++i) { a[i] = l[i]; a[i + 8] = h[i]; }
  return a;
}

static __device__ __forceinline__ unsigned lds_off(const void* p) {
  // generic LDS pointer: addr[31:0] is the LDS byte offset (ISA 10.2 aperture rules)
  return (unsigned)(uintptr_t)p;
}

// ---------------- weight conversion ----------------
__global__ __launch_bounds__(256) void k_convert(const float* __restrict__ wq,
                                                 const float* __restrict__ wf,
                                                 f16* __restrict__ wqh,
                                                 f16* __restrict__ wfh) {
  int i = blockIdx.x * 256 + threadIdx.x;
  if (i < 384 * 1152) wqh[i] = (f16)wq[i];
  if (i < 384 * 384)  wfh[i] = (f16)wf[i];
}

// ---------------- QKV GEMM: [100352,384] x [384,1152], de-interleave + shift(-4) ----------------
__global__ __launch_bounds__(256) void k_qkv(const float* __restrict__ x,
                                             const f16* __restrict__ wqh,
                                             const float* __restrict__ bq,
                                             f16* __restrict__ qws,
                                             f16* __restrict__ kws,
                                             f16* __restrict__ vws) {
  __shared__ f16 at[128 * 32];
  __shared__ f16 bt[64 * 32];   // transposed: bt[col][k]
  const int tid = threadIdx.x;
  const int lane = tid & 31, wave = tid >> 5;
  const int hf = lane >> 4, ln = lane & 15;
  const int wm = wave & 3, wn = wave >> 2;
  const int rowBase = blockIdx.x * 128;
  const int colBase = blockIdx.y * 64;
  v8f acc[2][2] = {};
  for (int ks = 0; ks < 12; ++ks) {
#pragma unroll
    for (int t = 0; t < 16; ++t) {            // stage A: 128x32, f32->f16 (conversion => no DMA)
      int e = tid + t * 256;
      int r = e >> 5, c = e & 31;
      at[e] = (f16)x[(size_t)(rowBase + r) * 384 + ks * 32 + c];
    }
#pragma unroll
    for (int t = 0; t < 8; ++t) {             // stage B transposed: 32x64 -> bt[64][32]
      int e = tid + t * 256;
      int kr = e >> 6, n = e & 63;
      bt[n * 32 + kr] = wqh[(size_t)(ks * 32 + kr) * 1152 + colBase + n];
    }
    __syncthreads();
    if (ks + 1 < 12)
      __builtin_prefetch(&x[(size_t)(rowBase + (tid >> 1)) * 384 + (ks + 1) * 32], 0, 1);
    const int ra = wm * 32 + ln;
    v16h a0 = afrag(&at[ra * 32 + 16 * hf], &at[ra * 32 + 8 + 16 * hf]);
    v16h a1 = afrag(&at[(ra + 16) * 32 + 16 * hf], &at[(ra + 16) * 32 + 8 + 16 * hf]);
    const int cb = wn * 32 + ln;
    v16h b0 = *(const v16h*)&bt[cb * 32 + 16 * hf];
    v16h b1 = *(const v16h*)&bt[(cb + 16) * 32 + 16 * hf];
    acc[0][0] = wmma_f16(a0, b0, acc[0][0]);
    acc[0][1] = wmma_f16(a0, b1, acc[0][1]);
    acc[1][0] = wmma_f16(a1, b0, acc[1][0]);
    acc[1][1] = wmma_f16(a1, b1, acc[1][1]);
    __syncthreads();
  }
  // epilogue: +bias, split (c,3) columns, write shifted (-4,-4) f16 q/k/v
#pragma unroll
  for (int i = 0; i < 2; ++i)
#pragma unroll
    for (int j = 0; j < 2; ++j)
#pragma unroll
      for (int r = 0; r < 8; ++r) {
        int M = rowBase + wm * 32 + i * 16 + r + 8 * hf;
        int J = colBase + wn * 32 + j * 16 + ln;
        float val = acc[i][j][r] + bq[J];
        int bb = M / 3136, n = M - bb * 3136;
        int h = n / 56, w = n - h * 56;
        int hs = (h >= 4) ? h - 4 : h + 52;
        int wsft = (w >= 4) ? w - 4 : w + 52;
        int c = J / 3, which = J - c * 3;
        size_t idx = (size_t)(bb * 3136 + hs * 56 + wsft) * 384 + c;
        f16* p = (which == 0) ? qws : (which == 1) ? kws : vws;
        p[idx] = (f16)val;
      }
}

// ---------------- windowed attention: one wave per (b, head, window) ----------------
__global__ __launch_bounds__(256) void k_attn(const f16* __restrict__ qws,
                                              const f16* __restrict__ kws,
                                              const f16* __restrict__ vws,
                                              const float* __restrict__ rb,
                                              f16* __restrict__ ows) {
  __shared__ f16 plds[8][16 * 64];            // per-wave P tile (16 rows x 64 keys)
  __shared__ f16 vstage[8][64 * 32];          // per-wave V tile (64 keys x 32 ch), async-staged
  const int tid = threadIdx.x;
  const int lane = tid & 31, wv = tid >> 5;   // wave id == window column
  const int hf = lane >> 4, ln = lane & 15;
  const int wu = blockIdx.x, head = blockIdx.y, b = blockIdx.z;
  const size_t base = (size_t)b * 3136 * 384 + head * 32;
  f16* P = plds[wv];
  f16* VS = vstage[wv];
  const float scale = 0.17677669529663687f;   // 1/sqrt(32)

  // ---- async-stage V window rows into LDS (ASYNCcnt path), 16B per lane per op ----
#pragma unroll
  for (int t = 0; t < 8; ++t) {
    int e = t * 32 + lane;                    // 0..255 -> row 0..63, 16B chunk 0..3
    int row = e >> 2;
    int ch = (e & 3) * 8;
    if (row < 49) {
      int i = row / 7, j = row - i * 7;
      const f16* g = &vws[base + (size_t)((wu * 7 + i) * 56 + (wv * 7 + j)) * 384 + ch];
      unsigned l = lds_off(&VS[row * 32 + ch]);
      asm volatile("global_load_async_to_lds_b128 %0, %1, off"
                   :: "v"(l), "v"(g) : "memory");
    }
  }

  // K fragments (B-layout): lane col = key token, elems = channel 16*hf..+15
  v16h kf[4];
#pragma unroll
  for (int nt = 0; nt < 4; ++nt) {
    int key = nt * 16 + ln;
    if (key < 49) {
      int i = key / 7, j = key - i * 7;
      int row = (wu * 7 + i) * 56 + (wv * 7 + j);
      kf[nt] = *(const v16h*)&kws[base + (size_t)row * 384 + 16 * hf];
    } else {
      v16h z = {}; kf[nt] = z;
    }
  }

  // wait for async V staging, then build V fragments from LDS
  asm volatile("s_wait_asynccnt 0x0" ::: "memory");
  v16h vf[2][2];
#pragma unroll
  for (int kt = 0; kt < 2; ++kt)
#pragma unroll
    for (int ct = 0; ct < 2; ++ct) {
      v16h t = {};
#pragma unroll
      for (int e = 0; e < 16; ++e) {
        int key = kt * 32 + e + 16 * hf;
        if (key < 49) t[e] = VS[key * 32 + ct * 16 + ln];
      }
      vf[kt][ct] = t;
    }

  for (int mt = 0; mt < 4; ++mt) {
    int qt0 = mt * 16 + ln;
    v16h qa;
    if (qt0 < 49) {
      int i = qt0 / 7, j = qt0 - i * 7;
      size_t ro = base + (size_t)((wu * 7 + i) * 56 + (wv * 7 + j)) * 384;
      qa = afrag(&qws[ro + 16 * hf], &qws[ro + 8 + 16 * hf]);
    } else {
      v16h z = {}; qa = z;
    }
    v8f z8 = {};
    v8f s[4];
#pragma unroll
    for (int nt = 0; nt < 4; ++nt) s[nt] = wmma_f16(qa, kf[nt], z8);

    // scale + relative position bias + shift masks
#pragma unroll
    for (int nt = 0; nt < 4; ++nt) {
      int kc = nt * 16 + ln;
#pragma unroll
      for (int r = 0; r < 8; ++r) {
        int qt = mt * 16 + r + 8 * hf;
        float sv = s[nt][r] * scale;
        if (kc >= 49) {
          sv = -__builtin_inff();
        } else if (qt < 49) {
          sv += rb[qt * 49 + kc];
          if (wu == 7 && ((qt / 7 >= 4) != (kc / 7 >= 4))) sv = -__builtin_inff();
          if (wv == 7 && ((qt % 7 >= 4) != (kc % 7 >= 4))) sv = -__builtin_inff();
        }
        s[nt][r] = sv;
      }
    }
    // row softmax (rows live across 16-lane halves; xor 1,2,4,8 stays in-half)
#pragma unroll
    for (int r = 0; r < 8; ++r) {
      float m = fmaxf(fmaxf(s[0][r], s[1][r]), fmaxf(s[2][r], s[3][r]));
      m = fmaxf(m, __shfl_xor(m, 1, 32));
      m = fmaxf(m, __shfl_xor(m, 2, 32));
      m = fmaxf(m, __shfl_xor(m, 4, 32));
      m = fmaxf(m, __shfl_xor(m, 8, 32));
      float e0 = __expf(s[0][r] - m), e1 = __expf(s[1][r] - m);
      float e2 = __expf(s[2][r] - m), e3 = __expf(s[3][r] - m);
      float sm = e0 + e1 + e2 + e3;
      sm += __shfl_xor(sm, 1, 32);
      sm += __shfl_xor(sm, 2, 32);
      sm += __shfl_xor(sm, 4, 32);
      sm += __shfl_xor(sm, 8, 32);
      float inv = 1.0f / sm;
      int lr = r + 8 * hf;
      P[lr * 64 +      ln] = (f16)(e0 * inv);
      P[lr * 64 + 16 + ln] = (f16)(e1 * inv);
      P[lr * 64 + 32 + ln] = (f16)(e2 * inv);
      P[lr * 64 + 48 + ln] = (f16)(e3 * inv);
    }
    __syncthreads();
    v16h pa0 = afrag(&P[ln * 64 + 16 * hf], &P[ln * 64 + 8 + 16 * hf]);
    v16h pa1 = afrag(&P[ln * 64 + 32 + 16 * hf], &P[ln * 64 + 40 + 16 * hf]);
#pragma unroll
    for (int ct = 0; ct < 2; ++ct) {
      v8f oacc = wmma_f16(pa0, vf[0][ct], z8);
      oacc = wmma_f16(pa1, vf[1][ct], oacc);
#pragma unroll
      for (int r = 0; r < 8; ++r) {
        int tok = mt * 16 + r + 8 * hf;
        if (tok < 49) {
          int i = tok / 7, j = tok - i * 7;
          int gh = wu * 7 + i, gw = wv * 7 + j;
          int fh = gh + 3; if (fh >= 56) fh -= 56;   // reverse roll +3 (asymmetric, per source)
          int fw = gw + 3; if (fw >= 56) fw -= 56;
          ows[(size_t)(b * 3136 + fh * 56 + fw) * 384 + head * 32 + ct * 16 + ln] = (f16)oacc[r];
        }
      }
    }
    __syncthreads();
  }
}

// ---------------- FFD GEMM: [100352,384] x [384,384] + bias -> f32 out ----------------
// A tile (128x32 f16, row stride 384) is moved by the Tensor Data Mover.
__global__ __launch_bounds__(256) void k_ffd(const f16* __restrict__ ows,
                                             const f16* __restrict__ wfh,
                                             const float* __restrict__ bf,
                                             float* __restrict__ out) {
  __shared__ f16 at[128 * 32];
  __shared__ f16 bt[64 * 32];
  const int tid = threadIdx.x;
  const int lane = tid & 31, wave = tid >> 5;
  const int hf = lane >> 4, ln = lane & 15;
  const int wm = wave & 3, wn = wave >> 2;
  const int rowBase = blockIdx.x * 128;
  const int colBase = blockIdx.y * 64;
  v8f acc[2][2] = {};
  for (int ks = 0; ks < 12; ++ks) {
    if (wave == 0) {
      // D# per ISA 8.3/8.4: count=1, type=2("image"), data_size=2B,
      // tile 32(x) x 128(y), tensor_dim0=384, dim0_stride=384, dims 2..4 trivial.
      uint64_t ga = (uint64_t)(uintptr_t)&ows[(size_t)rowBase * 384 + ks * 32];
      unsigned la = lds_off(&at[0]);
      v4u g0 = { 1u, la, (unsigned)ga, (unsigned)(ga >> 32) | 0x80000000u };
      v8i g1 = { (int)0x00010000,        // data_size=1 (2 bytes)
                 (int)0x01800000,        // tensor_dim0[15:0]=384 in bits[31:16]
                 (int)0x88000000u,       // tensor_dim1[15:0]=0x8800 (100352) in bits[31:16]
                 (int)0x00200001,        // tensor_dim1[31:16]=1, tile_dim0=32
                 (int)128,               // tile_dim1=128, tile_dim2=0
                 (int)384,               // tensor_dim0_stride low
                 (int)0x01800000,        // tensor_dim1_stride[15:0]=384 in bits[31:16]
                 0 };
      v4i g2 = { 1, 1, 0, 0 };           // tensor_dim2=1, tensor_dim3=1, tile_dim3=0
      v4i g3 = { 0, (int)0x00010000, 0, 0 }; // tensor_dim4=1, tile_dim4=0
      v8i g4 = { 0, 0, 0, 0, 0, 0, 0, 0 };   // extra group (6-arg builtin form), unused
      __builtin_amdgcn_tensor_load_to_lds(g0, g1, g2, g3, g4, 0);
      __builtin_amdgcn_s_wait_tensorcnt(0);
    }
#pragma unroll
    for (int t = 0; t < 8; ++t) {        // stage B transposed: 32x64 -> bt[64][32]
      int e = tid + t * 256;
      int kr = e >> 6, n = e & 63;
      bt[n * 32 + kr] = wfh[(size_t)(ks * 32 + kr) * 384 + colBase + n];
    }
    __syncthreads();
    if (ks + 1 < 12)
      __builtin_prefetch(&wfh[(size_t)((ks + 1) * 32 + (tid >> 3)) * 384 + colBase], 0, 1);
    const int ra = wm * 32 + ln;
    v16h a0 = afrag(&at[ra * 32 + 16 * hf], &at[ra * 32 + 8 + 16 * hf]);
    v16h a1 = afrag(&at[(ra + 16) * 32 + 16 * hf], &at[(ra + 16) * 32 + 8 + 16 * hf]);
    const int cb = wn * 32 + ln;
    v16h b0 = *(const v16h*)&bt[cb * 32 + 16 * hf];
    v16h b1 = *(const v16h*)&bt[(cb + 16) * 32 + 16 * hf];
    acc[0][0] = wmma_f16(a0, b0, acc[0][0]);
    acc[0][1] = wmma_f16(a0, b1, acc[0][1]);
    acc[1][0] = wmma_f16(a1, b0, acc[1][0]);
    acc[1][1] = wmma_f16(a1, b1, acc[1][1]);
    __syncthreads();
  }
#pragma unroll
  for (int i = 0; i < 2; ++i)
#pragma unroll
    for (int j = 0; j < 2; ++j)
#pragma unroll
      for (int r = 0; r < 8; ++r) {
        int M = rowBase + wm * 32 + i * 16 + r + 8 * hf;
        int J = colBase + wn * 32 + j * 16 + ln;
        out[(size_t)M * 384 + J] = acc[i][j][r] + bf[J];
      }
}

extern "C" void kernel_launch(void* const* d_in, const int* in_sizes, int n_in,
                              void* d_out, int out_size, void* d_ws, size_t ws_size,
                              hipStream_t stream) {
  const float* x      = (const float*)d_in[0];
  const float* w_qkv  = (const float*)d_in[1];
  const float* b_qkv  = (const float*)d_in[2];
  const float* w_ffd  = (const float*)d_in[3];
  const float* b_ffd  = (const float*)d_in[4];
  const float* relb   = (const float*)d_in[5];

  const size_t NE = (size_t)32 * 3136 * 384;  // elements per q/k/v/o buffer
  f16* wqh = (f16*)d_ws;
  f16* wfh = wqh + (size_t)384 * 1152;
  f16* qws = wfh + (size_t)384 * 384;
  f16* kws = qws + NE;
  f16* vws = kws + NE;
  f16* ows = vws + NE;

  k_convert<<<1728, 256, 0, stream>>>(w_qkv, w_ffd, wqh, wfh);
  k_qkv<<<dim3(784, 18), 256, 0, stream>>>(x, wqh, b_qkv, qws, kws, vws);
  k_attn<<<dim3(8, 12, 32), 256, 0, stream>>>(qws, kws, vws, relb, ows);
  k_ffd<<<dim3(784, 6), 256, 0, stream>>>(ows, wfh, b_ffd, (float*)d_out);
}
